// Patches_Atten1_86938728006216
// MI455X (gfx1250) — matrix operand order repaired and checked by reading, not compiled, not run on hardware
//
#include <hip/hip_runtime.h>
#include <math.h>

typedef __attribute__((ext_vector_type(16))) _Float16 v16h;
typedef __attribute__((ext_vector_type(8)))  _Float16 v8h;
typedef __attribute__((ext_vector_type(8)))  float    v8f;
typedef __attribute__((ext_vector_type(4)))  int      v4i;

#define ZV8 ((v8f){0.f,0.f,0.f,0.f,0.f,0.f,0.f,0.f})

#if defined(__gfx1250__) && __has_builtin(__builtin_amdgcn_global_load_async_to_lds_b128)
#define HAVE_ASYNC_LDS 1
#else
#define HAVE_ASYNC_LDS 0
#endif

#if HAVE_ASYNC_LDS
typedef __attribute__((address_space(1))) v4i* gas_v4i;
typedef __attribute__((address_space(3))) v4i* las_v4i;
#endif

// Copy 8 fp16 (16 bytes) global -> LDS. Async path uses the CDNA5 direct
// global-to-LDS DMA (tracked by ASYNCcnt); fallback is a vector copy.
__device__ __forceinline__ void cp16(const _Float16* g, _Float16* l) {
#if HAVE_ASYNC_LDS
  __builtin_amdgcn_global_load_async_to_lds_b128((gas_v4i)(void*)g,
                                                 (las_v4i)(void*)l, 0, 0);
#else
  *(v8h*)l = *(const v8h*)g;
#endif
}

__device__ __forceinline__ void wait_async_lds() {
#if HAVE_ASYNC_LDS
  asm volatile("s_wait_asynccnt 0x0" ::: "memory");
#endif
}

__device__ __forceinline__ v8f wmma16(v16h a, v16h b, v8f c) {
  return __builtin_amdgcn_wmma_f32_16x16x32_f16(false, a, false, b, (short)0, c,
                                                false, false);
}

// A fragment 16x32 f16 from LDS, row-major [m][32] (lda fixed 32, 16B-aligned).
// lane L: M=L&15; lanes<16 hold K={0..15}, lanes>=16 K={16..31}.
__device__ __forceinline__ v16h frag_a_lds32(const _Float16* p) {
  const int lane = threadIdx.x & 31;
  const _Float16* row = p + ((lane & 15) << 5) + ((lane >> 4) << 4);
  v8h lo = *(const v8h*)row;
  v8h hi = *(const v8h*)(row + 8);
  return __builtin_shufflevector(lo, hi, 0, 1, 2, 3, 4, 5, 6, 7, 8, 9, 10, 11,
                                 12, 13, 14, 15);
}

// B fragment from a swizzled 16-column panel: lane L's 16 halves are contiguous.
// Stored element (k, n) of the panel lives at lane=((k>>4)<<4)|(n&15), j=k&15.
__device__ __forceinline__ v16h frag_b_swiz(const _Float16* panel) {
  return *(const v16h*)(panel + ((threadIdx.x & 31) << 4));
}

// swizzled offset inside a 4-panel (64-column, 32-k) B tile
__device__ __forceinline__ int bswz(int k, int n) {
  return ((n >> 4) << 9) + (((((k >> 4) << 4)) | (n & 15)) << 4) + (k & 15);
}

// ---------------------------------------------------------------------------
// Kernel 1: fused QKV 1x1-conv GEMM + patchify to token-major per-scale fp16.
// grid (64 hw-tiles, 4 o-tiles(=scale), 16 images), block 256.
// ---------------------------------------------------------------------------
__global__ __launch_bounds__(256)
void qkv_project_kernel(const float* __restrict__ xs,
                        const float* __restrict__ Wq, const float* __restrict__ bq,
                        const float* __restrict__ Wk, const float* __restrict__ bk,
                        const float* __restrict__ Wv, const float* __restrict__ bv,
                        _Float16* __restrict__ Qt, _Float16* __restrict__ Kt,
                        _Float16* __restrict__ Vt) {
  __shared__ __attribute__((aligned(64))) _Float16 lxs[2048];      // swizzled X tile
  __shared__ __attribute__((aligned(64))) _Float16 lw[3][64][32];  // weights [qkv][o][k]
  const int n = blockIdx.z, o0 = blockIdx.y * 64, hw0 = blockIdx.x * 64;
  const int tid = threadIdx.x, lane = tid & 31, wv = tid >> 5;

  v8f acc[6];
#pragma unroll
  for (int t = 0; t < 6; ++t) acc[t] = ZV8;

  for (int kc = 0; kc < 256; kc += 32) {
    for (int idx = tid; idx < 2048; idx += 256) {
      int kk = idx >> 6, j = idx & 63;
      float xv = xs[((size_t)n * 256 + kc + kk) * 4096 + hw0 + j];
      lxs[bswz(kk, j)] = (_Float16)xv;
    }
    for (int idx = tid; idx < 6144; idx += 256) {
      int e = idx >> 11, rem = idx & 2047;
      int oo = rem >> 5, kk = rem & 31;
      const float* We = (e == 0) ? Wq : ((e == 1) ? Wk : Wv);
      lw[e][oo][kk] = (_Float16)We[(o0 + oo) * 256 + kc + kk];
    }
    __syncthreads();
    if (kc + 32 < 256)
      __builtin_prefetch(xs + ((size_t)n * 256 + kc + 32 + (tid >> 3)) * 4096 + hw0, 0, 1);
#pragma unroll
    for (int t6 = 0; t6 < 6; ++t6) {
      int tileid = wv * 6 + t6;
      int e = tileid >> 4, rem = tileid & 15;
      int otr = rem >> 2, ntc = rem & 3;
      v16h a = frag_a_lds32(&lw[e][otr * 16][0]);
      v16h b = frag_b_swiz(lxs + (ntc << 9));
      acc[t6] = wmma16(a, b, acc[t6]);
    }
    __syncthreads();
  }

  const int i = blockIdx.y;  // scale index (d_k = 64)
  const int lp = 5 - i, lo = 6 - lp, dsh = 6 + 2 * lp;
  const int pm = (1 << lp) - 1;
  const int bb = n >> 3, tt = n & 7;
  const int nn = lane & 15, hh = lane >> 4;
#pragma unroll
  for (int t6 = 0; t6 < 6; ++t6) {
    int tileid = wv * 6 + t6;
    int e = tileid >> 4, rem = tileid & 15;
    int otr = rem >> 2, ntc = rem & 3;
    _Float16* op = (e == 0) ? Qt : ((e == 1) ? Kt : Vt);
    const float* bp = (e == 0) ? bq : ((e == 1) ? bk : bv);
#pragma unroll
    for (int r = 0; r < 8; ++r) {
      int o = o0 + otr * 16 + r + 8 * hh;
      int c = o & 63;
      int hwp = hw0 + ntc * 16 + nn;
      int hy = hwp >> 6, hx = hwp & 63;
      int oy = hy >> lp, py = hy & pm, ox = hx >> lp, px = hx & pm;
      int tok = (((tt << lo) | oy) << lo) | ox;
      int f = (c << (2 * lp)) | (py << lp) | px;
      size_t addr = ((size_t)(i * 2 + bb) << 21) + ((size_t)tok << dsh) + f;
      op[addr] = (_Float16)(acc[t6][r] + bp[o]);
    }
  }
}

// ---------------------------------------------------------------------------
// Kernel 2: S = Q K^T / sqrt(d). grid (mt*nt, 2 batches), block 256.
// ---------------------------------------------------------------------------
__global__ __launch_bounds__(256)
void scores_kernel(const _Float16* __restrict__ Qt, const _Float16* __restrict__ Kt,
                   float* __restrict__ S, int scale, int N, int d, int nt, float sc) {
  __shared__ __attribute__((aligned(64))) _Float16 lq[64][32];  // A tile [m][k]
  __shared__ __attribute__((aligned(64))) _Float16 lks[2048];   // swizzled K^T tile
  const int bb = blockIdx.y;
  const _Float16* qb = Qt + ((size_t)(scale * 2 + bb) << 21);
  const _Float16* kb = Kt + ((size_t)(scale * 2 + bb) << 21);
  const int m0 = (blockIdx.x / nt) * 64, n0 = (blockIdx.x % nt) * 64;
  const int tid = threadIdx.x, lane = tid & 31, wv = tid >> 5;

  v8f acc[2];
  acc[0] = ZV8; acc[1] = ZV8;

  for (int f0 = 0; f0 < d; f0 += 32) {
    {  // A tile: 256 x 16-byte chunks, one per thread (async DMA to LDS)
      int mm = tid >> 2, k8 = (tid & 3) << 3;
      int row = m0 + mm;
      _Float16* dst = &lq[mm][k8];
      if (row < N) cp16(qb + (size_t)row * d + f0 + k8, dst);
      else { v8h z = {}; *(v8h*)dst = z; }
    }
    {  // B tile (K^T), swizzled: both sides contiguous along k -> async b128
      int nc = tid >> 2, k8 = (tid & 3) << 3;
      int col = n0 + nc;
      _Float16* dst = lks + bswz(k8, nc);
      if (col < N) cp16(kb + (size_t)col * d + f0 + k8, dst);
      else { v8h z = {}; *(v8h*)dst = z; }
    }
    wait_async_lds();
    __syncthreads();
    if (f0 + 32 < d)
      __builtin_prefetch(qb + (size_t)(m0 + (tid >> 2)) * d + f0 + 32, 0, 1);
#pragma unroll
    for (int t = 0; t < 2; ++t) {
      int tileid = wv * 2 + t;
      int otr = tileid >> 2, ntc = tileid & 3;
      v16h a = frag_a_lds32(&lq[otr * 16][0]);
      v16h b = frag_b_swiz(lks + (ntc << 9));
      acc[t] = wmma16(a, b, acc[t]);
    }
    __syncthreads();
  }

  const int nn = lane & 15, hh = lane >> 4;
#pragma unroll
  for (int t = 0; t < 2; ++t) {
    int tileid = wv * 2 + t;
    int otr = tileid >> 2, ntc = tileid & 3;
#pragma unroll
    for (int r = 0; r < 8; ++r) {
      int m = m0 + otr * 16 + r + 8 * hh;
      int nc = n0 + ntc * 16 + nn;
      if (m < N && nc < N)
        S[(size_t)bb * N * N + (size_t)m * N + nc] = acc[t][r] * sc;
    }
  }
}

// ---------------------------------------------------------------------------
// Kernel 3: row softmax, fp32 in -> fp16 probabilities out. grid (N, 2).
// ---------------------------------------------------------------------------
__global__ __launch_bounds__(256)
void softmax_kernel(const float* __restrict__ S, _Float16* __restrict__ P, int N) {
  const int row = blockIdx.x, bb = blockIdx.y, tid = threadIdx.x;
  const float* s = S + (size_t)bb * N * N + (size_t)row * N;
  _Float16* p = P + (size_t)bb * N * N + (size_t)row * N;
  __shared__ float red[256];

  float v[8];
  float mx = -3.0e38f;
#pragma unroll
  for (int j = 0; j < 8; ++j) {
    int idx = tid + j * 256;
    v[j] = (idx < N) ? s[idx] : -3.0e38f;
    mx = fmaxf(mx, v[j]);
  }
  red[tid] = mx; __syncthreads();
  for (int st = 128; st > 0; st >>= 1) {
    if (tid < st) red[tid] = fmaxf(red[tid], red[tid + st]);
    __syncthreads();
  }
  mx = red[0]; __syncthreads();

  float sum = 0.f;
#pragma unroll
  for (int j = 0; j < 8; ++j) {
    int idx = tid + j * 256;
    v[j] = (idx < N) ? __expf(v[j] - mx) : 0.f;
    sum += v[j];
  }
  red[tid] = sum; __syncthreads();
  for (int st = 128; st > 0; st >>= 1) {
    if (tid < st) red[tid] += red[tid + st];
    __syncthreads();
  }
  float inv = 1.f / red[0];
#pragma unroll
  for (int j = 0; j < 8; ++j) {
    int idx = tid + j * 256;
    if (idx < N) p[idx] = (_Float16)(v[j] * inv);
  }
}

// ---------------------------------------------------------------------------
// Kernel 4: O = P V, epilogue de-patchifies into NCHW fp16 atten_res.
// grid (nt*ft, 2), block 256.
// ---------------------------------------------------------------------------
__global__ __launch_bounds__(256)
void av_kernel(const _Float16* __restrict__ P, const _Float16* __restrict__ Vt,
               _Float16* __restrict__ att, int scale, int N, int d, int ft, int lp) {
  __shared__ __attribute__((aligned(64))) _Float16 lpA[64][32];  // P tile [m][k]
  __shared__ __attribute__((aligned(64))) _Float16 lvs[2048];    // swizzled V tile
  const int bb = blockIdx.y;
  const int m0 = (blockIdx.x / ft) * 64, f0 = (blockIdx.x % ft) * 64;
  const _Float16* pb = P + (size_t)bb * N * N;
  const _Float16* vb = Vt + ((size_t)(scale * 2 + bb) << 21);
  const int tid = threadIdx.x, lane = tid & 31, wv = tid >> 5;

  v8f acc[2];
  acc[0] = ZV8; acc[1] = ZV8;

  for (int k0 = 0; k0 < N; k0 += 32) {
    {  // A tile (P): async b128 copies
      int mm = tid >> 2, k8 = (tid & 3) << 3;
      int row = m0 + mm;
      _Float16* dst = &lpA[mm][k8];
      if (row < N) cp16(pb + (size_t)row * N + k0 + k8, dst);
      else { v8h z = {}; *(v8h*)dst = z; }
    }
    // B tile (V): transpose-swizzle, global-coalesced reads, strided LDS writes
    for (int idx = tid; idx < 2048; idx += 256) {
      int kk = idx >> 6, ff = idx & 63;
      lvs[bswz(kk, ff)] = vb[(size_t)(k0 + kk) * d + f0 + ff];
    }
    wait_async_lds();
    __syncthreads();
    if (k0 + 32 < N)
      __builtin_prefetch(vb + (size_t)(k0 + 32 + (tid >> 3)) * d + f0, 0, 1);
#pragma unroll
    for (int t = 0; t < 2; ++t) {
      int tileid = wv * 2 + t;
      int otr = tileid >> 2, ntc = tileid & 3;
      v16h a = frag_a_lds32(&lpA[otr * 16][0]);
      v16h b = frag_b_swiz(lvs + (ntc << 9));
      acc[t] = wmma16(a, b, acc[t]);
    }
    __syncthreads();
  }

  const int lo = 6 - lp, pm = (1 << lp) - 1, om = (1 << lo) - 1;
  const int nn = lane & 15, hh = lane >> 4;
#pragma unroll
  for (int t = 0; t < 2; ++t) {
    int tileid = wv * 2 + t;
    int otr = tileid >> 2, ntc = tileid & 3;
#pragma unroll
    for (int r = 0; r < 8; ++r) {
      int m = m0 + otr * 16 + r + 8 * hh;  // token
      int fv = f0 + ntc * 16 + nn;         // feature
      if (m < N) {
        int tt = m >> (2 * lo);
        int rem = m & ((1 << (2 * lo)) - 1);
        int oy = rem >> lo, ox = rem & om;
        int c = fv >> (2 * lp);
        int r2 = fv & ((1 << (2 * lp)) - 1);
        int py = r2 >> lp, px = r2 & pm;
        int o = scale * 64 + c;
        int img = bb * 8 + tt;
        int hy = (oy << lp) | py, hx = (ox << lp) | px;
        att[(((size_t)img * 256 + o) << 12) + (hy << 6) + hx] = (_Float16)acc[t][r];
      }
    }
  }
}

// ---------------------------------------------------------------------------
// Kernel 5: 3x3 conv (dilation 1 or 2, pad = dil) as implicit WMMA GEMM,
// fused leaky-ReLU + optional residual, dual fp32/fp16 output.
// Input patch stored pixel-major [pix][c] so B fragments are contiguous 32B.
// grid (16 spatial tiles, 4 o-tiles, 16 images), block 256.
// ---------------------------------------------------------------------------
__global__ __launch_bounds__(256)
void conv3x3_kernel(const _Float16* __restrict__ xin, const float* __restrict__ Wc,
                    const float* __restrict__ bc, const float* __restrict__ res,
                    float* __restrict__ out32, _Float16* __restrict__ out16, int dil) {
  __shared__ __attribute__((aligned(64))) _Float16 lin[400 * 32];   // [20*20][c]
  __shared__ __attribute__((aligned(64))) _Float16 lwc[9][64][32];  // [tap][o][c]
  const int n = blockIdx.z, o0 = blockIdx.y * 64;
  const int ty0 = (blockIdx.x >> 2) * 16, tx0 = (blockIdx.x & 3) * 16;
  const int tid = threadIdx.x, lane = tid & 31, wv = tid >> 5;
  const int SDr = 16 + 2 * dil, S2 = SDr * SDr;

  v8f acc[8];
#pragma unroll
  for (int t = 0; t < 8; ++t) acc[t] = ZV8;

  for (int cc = 0; cc < 256; cc += 32) {
    __syncthreads();
    for (int idx = tid; idx < 32 * S2; idx += 256) {
      int cl = idx / S2, rem = idx % S2;
      int sy = rem / SDr, sx = rem % SDr;
      int y = ty0 + sy - dil, x = tx0 + sx - dil;
      _Float16 v = (_Float16)0.f;
      if (y >= 0 && y < 64 && x >= 0 && x < 64)
        v = xin[((size_t)n * 256 + cc + cl) * 4096 + (y << 6) + x];
      lin[(sy * 20 + sx) * 32 + cl] = v;
    }
    for (int idx = tid; idx < 18432; idx += 256) {
      int tap = idx >> 11, rem = idx & 2047;
      int oo = rem >> 5, cl = rem & 31;
      lwc[tap][oo][cl] = (_Float16)Wc[((size_t)(o0 + oo) * 256 + cc + cl) * 9 + tap];
    }
    __syncthreads();
    const int nn = lane & 15, hh = lane >> 4;
    for (int tap = 0; tap < 9; ++tap) {
      int ky = tap / 3, kx = tap % 3;
#pragma unroll
      for (int t = 0; t < 8; ++t) {
        int otr = t >> 1, ctl = wv * 2 + (t & 1);
        v16h a = frag_a_lds32(&lwc[tap][otr * 16][0]);
        int p = ctl * 16 + nn;
        const _Float16* bp =
            lin + (((p >> 4) + ky * dil) * 20 + ((p & 15) + kx * dil)) * 32 + hh * 16;
        v16h b = *(const v16h*)bp;
        acc[t] = wmma16(a, b, acc[t]);
      }
    }
  }

  const int nn = lane & 15, hh = lane >> 4;
#pragma unroll
  for (int t = 0; t < 8; ++t) {
    int otr = t >> 1, ctl = wv * 2 + (t & 1);
#pragma unroll
    for (int r = 0; r < 8; ++r) {
      int o = o0 + otr * 16 + r + 8 * hh;
      int p = ctl * 16 + nn;
      int y = ty0 + (p >> 4), x = tx0 + (p & 15);
      float v = acc[t][r] + bc[o];
      v = v > 0.f ? v : 0.2f * v;
      size_t idx = (((size_t)n * 256 + o) << 12) + (y << 6) + x;
      if (res) v += res[idx];
      if (out32) out32[idx] = v;
      if (out16) out16[idx] = (_Float16)v;
    }
  }
}

// ---------------------------------------------------------------------------
extern "C" void kernel_launch(void* const* d_in, const int* in_sizes, int n_in,
                              void* d_out, int out_size, void* d_ws, size_t ws_size,
                              hipStream_t stream) {
  (void)in_sizes; (void)n_in; (void)out_size; (void)ws_size;
  const float* xs  = (const float*)d_in[0];
  const float* Wq  = (const float*)d_in[2];
  const float* bq  = (const float*)d_in[3];
  const float* Wk  = (const float*)d_in[4];
  const float* bk  = (const float*)d_in[5];
  const float* Wv  = (const float*)d_in[6];
  const float* bv  = (const float*)d_in[7];
  const float* Wl  = (const float*)d_in[8];
  const float* bl  = (const float*)d_in[9];
  const float* Wf1 = (const float*)d_in[10];
  const float* bf1 = (const float*)d_in[11];
  const float* Wf2 = (const float*)d_in[12];
  const float* bf2 = (const float*)d_in[13];

  char* w = (char*)d_ws;
  _Float16* Qt  = (_Float16*)w; w += (size_t)32 << 20;
  _Float16* Kt  = (_Float16*)w; w += (size_t)32 << 20;
  _Float16* Vt  = (_Float16*)w; w += (size_t)32 << 20;
  float*    S   = (float*)w;    w += (size_t)32 << 20;
  _Float16* P   = (_Float16*)w; w += (size_t)16 << 20;
  _Float16* att = (_Float16*)w; w += (size_t)32 << 20;
  float*    x1  = (float*)w;    w += (size_t)64 << 20;
  _Float16* x1h = (_Float16*)w; w += (size_t)32 << 20;
  _Float16* ffh = (_Float16*)w;

  qkv_project_kernel<<<dim3(64, 4, 16), 256, 0, stream>>>(xs, Wq, bq, Wk, bk, Wv, bv,
                                                          Qt, Kt, Vt);

  const int Ns[4]  = {32, 128, 512, 2048};
  const int dsz[4] = {65536, 16384, 4096, 1024};
  const int lps[4] = {5, 4, 3, 2};
  for (int i = 0; i < 4; ++i) {
    int N = Ns[i], d = dsz[i];
    int nt = (N + 63) / 64;
    float sc = 1.0f / sqrtf((float)d);
    scores_kernel<<<dim3(nt * nt, 2), 256, 0, stream>>>(Qt, Kt, S, i, N, d, nt, sc);
    softmax_kernel<<<dim3(N, 2), 256, 0, stream>>>(S, P, N);
    int ft = d / 64;
    av_kernel<<<dim3(nt * ft, 2), 256, 0, stream>>>(P, Vt, att, i, N, d, ft, lps[i]);
  }

  conv3x3_kernel<<<dim3(16, 4, 16), 256, 0, stream>>>(att, Wl, bl, xs, x1, x1h, 1);
  conv3x3_kernel<<<dim3(16, 4, 16), 256, 0, stream>>>(x1h, Wf1, bf1, nullptr, nullptr,
                                                      ffh, 1);
  conv3x3_kernel<<<dim3(16, 4, 16), 256, 0, stream>>>(ffh, Wf2, bf2, x1, (float*)d_out,
                                                      nullptr, 2);
}
